// MotionTransformer_88158498718387
// MI455X (gfx1250) — matrix operand order repaired and checked by reading, not compiled, not run on hardware
//
#include <hip/hip_runtime.h>
#include <math.h>

// ---------------------------------------------------------------------------
// MotionTransformer forward for MI455X (gfx1250, wave32, WMMA).
// One generic strided/batched bf16 WMMA GEMM (f32 accumulate) services every
// matmul; VALU kernels handle LN / softmax / FiLM / elementwise.
// ---------------------------------------------------------------------------

#define B_    128
#define T_    196
#define FIN_  263
#define D_    512
#define DT_   256
#define H_    8
#define FF_   2048
#define E_    2048
#define L_    8
#define NTXT_ 77
#define DH_   64
#define BT_   (B_ * T_)   // 25088
#define CH_   3136        // BT_/8 FFN row chunk

typedef __bf16   v16bf __attribute__((ext_vector_type(16)));
typedef float    v8f   __attribute__((ext_vector_type(8)));
typedef unsigned v8u   __attribute__((ext_vector_type(8)));

enum { ACT_NONE = 0, ACT_GELU = 1, ACT_SILU = 2 };

__device__ __forceinline__ unsigned bfround(float f) {
  unsigned u = __builtin_bit_cast(unsigned, f);
  return u + 0x7FFFu + ((u >> 16) & 1u);     // round-to-nearest-even to bf16
}
__device__ __forceinline__ unsigned pack2bf(float lo, float hi) {
  return (bfround(lo) >> 16) | (bfround(hi) & 0xFFFF0000u);
}
__device__ __forceinline__ __bf16 f2bf(float f) {
  unsigned short h = (unsigned short)(bfround(f) >> 16);
  return __builtin_bit_cast(__bf16, h);
}

__device__ __forceinline__ float silu_f(float x) { return x / (1.f + __expf(-x)); }
__device__ __forceinline__ float gelu_f(float x) { return 0.5f * x * (1.f + erff(x * 0.70710678118654752f)); }
__device__ __forceinline__ int iclampmax(int v, int hi) { return v < hi ? v : hi; }

__device__ __forceinline__ v16bf mkfrag(uint4 lo, uint4 hi) {
  v8u u = {lo.x, lo.y, lo.z, lo.w, hi.x, hi.y, hi.z, hi.w};
  return __builtin_bit_cast(v16bf, u);
}

// ---------------------------------------------------------------------------
// Generic batched strided GEMM:  C[z] = act(A[z] @ W[z] + bias) + add[z]
//   A (m,k) at A + z0*a_bs0 + z1*a_bs1 + m*a_rs + k*a_cs
//   W (k,n) at W + z0*w_bs0 + z1*w_bs1 + k*w_rs + n*w_cs
//   C (m,n) at C + z0*c_bs0 + z1*c_bs1 + m*c_rs + n   (cols contiguous)
//   z0 = blockIdx.z / zdiv, z1 = blockIdx.z % zdiv
// 64x64 tile / workgroup (8 waves, wave = 16x32), K step 64 (2 k-halves),
// fp32 global -> packed bf16 LDS -> 4x v_wmma_f32_16x16x32_bf16 per stage.
// ---------------------------------------------------------------------------
struct GemmP {
  const float* A; const float* W; const float* bias; const float* add; float* C;
  int M, N, K;
  long a_rs, a_cs, w_rs, w_cs, c_rs;
  long a_bs0, a_bs1, w_bs0, w_bs1, c_bs0, c_bs1;
  int zdiv, act;
};

#define BMt 64
#define BNt 64
#define BKt 64
#define LDW 36   // dwords per LDS row: 32 used (64 bf16) + 4 pad (144B, 16B-aligned)

__global__ __launch_bounds__(256) void gemm_wmma(GemmP p) {
  // As: A tile, row-major per M row, dword j = bf16 {k=2j, k=2j+1}
  // Bt: W tile TRANSPOSED, row-major per N col, dword j = bf16 {k=2j, k=2j+1}
  __shared__ __align__(16) unsigned As[BMt * LDW];
  __shared__ __align__(16) unsigned Bt[BNt * LDW];
  __bf16* Ah  = reinterpret_cast<__bf16*>(As);   // half stride 2*LDW per row
  __bf16* Bth = reinterpret_cast<__bf16*>(Bt);

  const int tid = threadIdx.x;
  const int z   = blockIdx.z;
  const int z0  = z / p.zdiv, z1 = z % p.zdiv;
  const float* Ab = p.A + (long)z0 * p.a_bs0 + (long)z1 * p.a_bs1;
  const float* Wb = p.W + (long)z0 * p.w_bs0 + (long)z1 * p.w_bs1;
  float*       Cb = p.C + (long)z0 * p.c_bs0 + (long)z1 * p.c_bs1;
  const float* Rb = p.add ? p.add + (long)z0 * p.c_bs0 + (long)z1 * p.c_bs1 : nullptr;

  const int m0 = blockIdx.y * BMt;
  const int n0 = blockIdx.x * BNt;
  const int wid = tid >> 5, lane = tid & 31;
  const int wm = wid & 3, wn = wid >> 2;
  const int lhi = lane >> 4, llo = lane & 15;

  const bool fullM = (m0 + BMt <= p.M);
  const bool fullN = (n0 + BNt <= p.N);
  const bool acont = (p.a_cs == 1);
  const bool wcont = (p.w_cs == 1);

  v8f c0 = {}; v8f c1 = {};

  for (int k0 = 0; k0 < p.K; k0 += BKt) {
    const bool fullK = (k0 + BKt <= p.K);

    // ---- stage A tile [64 x 64] ----
    if (acont && fullM && fullK) {
      // uniform fast path: 2048 k-pairs, 8 per thread, no guards
#pragma unroll
      for (int i = 0; i < 8; ++i) {
        int idx = tid + i * 256;
        int m = idx >> 5, kp = idx & 31;
        const float* s = Ab + (long)(m0 + m) * p.a_rs + (k0 + 2 * kp);
        As[m * LDW + kp] = pack2bf(s[0], s[1]);
      }
    } else {
      // branch-free guarded path: clamped address + select-0, rolled loop
#pragma unroll 1
      for (int idx = tid; idx < BMt * BKt; idx += 256) {
        int m = idx >> 6, kk = idx & 63;
        int gm = m0 + m, gk = k0 + kk;
        long ao = (long)iclampmax(gm, p.M - 1) * p.a_rs +
                  (long)iclampmax(gk, p.K - 1) * p.a_cs;
        float v = Ab[ao];
        v = (gm < p.M && gk < p.K) ? v : 0.f;
        Ah[m * (2 * LDW) + kk] = f2bf(v);
      }
    }
    // ---- stage W tile [64 x 64], transposed into Bt ----
    if (wcont && fullN && fullK) {
#pragma unroll
      for (int i = 0; i < 8; ++i) {
        int idx = tid + i * 256;
        int kp = idx >> 6, n = idx & 63;
        const float* s = Wb + (long)(k0 + 2 * kp) * p.w_rs + (n0 + n);
        Bt[n * LDW + kp] = pack2bf(s[0], s[p.w_rs]);
      }
    } else {
#pragma unroll 1
      for (int idx = tid; idx < BKt * BNt; idx += 256) {
        int kk = idx >> 6, n = idx & 63;
        int gk = k0 + kk, gn = n0 + n;
        long wo = (long)iclampmax(gk, p.K - 1) * p.w_rs +
                  (long)iclampmax(gn, p.N - 1) * p.w_cs;
        float v = Wb[wo];
        v = (gk < p.K && gn < p.N) ? v : 0.f;
        Bth[n * (2 * LDW) + kk] = f2bf(v);
      }
    }
    __syncthreads();

    // ---- fragments (wave32; contiguous k order shared by A and B) ----
    // A: lane -> M row; within k-half: e <-> K = lhi*16 + e (contiguous, matches B)
    const int am = (wm * 16 + llo) * LDW;
    uint4 a0 = *reinterpret_cast<const uint4*>(&As[am + lhi * 8]);
    uint4 a1 = *reinterpret_cast<const uint4*>(&As[am + lhi * 8 + 4]);
    uint4 a2 = *reinterpret_cast<const uint4*>(&As[am + 16 + lhi * 8]);
    uint4 a3 = *reinterpret_cast<const uint4*>(&As[am + 16 + lhi * 8 + 4]);
    v16bf alo = mkfrag(a0, a1);
    v16bf ahi = mkfrag(a2, a3);
    // B: lane -> N col; e <-> K = lhi*16 + e (per k-half) => contiguous 32B runs
    const int bn0 = (wn * 32 + llo) * LDW;
    const int bn1 = (wn * 32 + 16 + llo) * LDW;
    v16bf b0lo = mkfrag(*reinterpret_cast<const uint4*>(&Bt[bn0 + lhi * 8]),
                        *reinterpret_cast<const uint4*>(&Bt[bn0 + lhi * 8 + 4]));
    v16bf b0hi = mkfrag(*reinterpret_cast<const uint4*>(&Bt[bn0 + 16 + lhi * 8]),
                        *reinterpret_cast<const uint4*>(&Bt[bn0 + 16 + lhi * 8 + 4]));
    v16bf b1lo = mkfrag(*reinterpret_cast<const uint4*>(&Bt[bn1 + lhi * 8]),
                        *reinterpret_cast<const uint4*>(&Bt[bn1 + lhi * 8 + 4]));
    v16bf b1hi = mkfrag(*reinterpret_cast<const uint4*>(&Bt[bn1 + 16 + lhi * 8]),
                        *reinterpret_cast<const uint4*>(&Bt[bn1 + 16 + lhi * 8 + 4]));

    c0 = __builtin_amdgcn_wmma_f32_16x16x32_bf16(false, alo, false, b0lo, (short)0, c0, false, false);
    c0 = __builtin_amdgcn_wmma_f32_16x16x32_bf16(false, ahi, false, b0hi, (short)0, c0, false, false);
    c1 = __builtin_amdgcn_wmma_f32_16x16x32_bf16(false, alo, false, b1lo, (short)0, c1, false, false);
    c1 = __builtin_amdgcn_wmma_f32_16x16x32_bf16(false, ahi, false, b1hi, (short)0, c1, false, false);
    __syncthreads();
  }

  // ---- epilogue: C/D layout row = r + 8*(lane>=16), col = llo ----
  const int colb0 = n0 + wn * 32 + llo;
  const int colb1 = colb0 + 16;
  if (fullM && fullN) {
#pragma unroll
    for (int r = 0; r < 8; ++r) {
      int row = m0 + wm * 16 + r + (lhi << 3);
      float v0 = c0[r], v1 = c1[r];
      if (p.bias) { v0 += p.bias[colb0]; v1 += p.bias[colb1]; }
      if (p.act == ACT_GELU) { v0 = gelu_f(v0); v1 = gelu_f(v1); }
      else if (p.act == ACT_SILU) { v0 = silu_f(v0); v1 = silu_f(v1); }
      long rb = (long)row * p.c_rs;
      if (Rb) { v0 += Rb[rb + colb0]; v1 += Rb[rb + colb1]; }
      Cb[rb + colb0] = v0;
      Cb[rb + colb1] = v1;
    }
  } else {
#pragma unroll 1
    for (int r = 0; r < 8; ++r) {
      int row = m0 + wm * 16 + r + (lhi << 3);
      if (row >= p.M) continue;
      float v0 = c0[r], v1 = c1[r];
      if (p.bias) {
        if (colb0 < p.N) v0 += p.bias[colb0];
        if (colb1 < p.N) v1 += p.bias[colb1];
      }
      if (p.act == ACT_GELU) { v0 = gelu_f(v0); v1 = gelu_f(v1); }
      else if (p.act == ACT_SILU) { v0 = silu_f(v0); v1 = silu_f(v1); }
      long rb = (long)row * p.c_rs;
      if (Rb) {
        if (colb0 < p.N) v0 += Rb[rb + colb0];
        if (colb1 < p.N) v1 += Rb[rb + colb1];
      }
      if (colb0 < p.N) Cb[rb + colb0] = v0;
      if (colb1 < p.N) Cb[rb + colb1] = v1;
    }
  }
}

// ---------------------------------------------------------------------------
// Elementwise / reduction kernels
// ---------------------------------------------------------------------------

// Sinusoidal timestep embedding: temb[b, 0:256]=cos, [256:512]=sin
__global__ void k_temb(const int* ts, float* temb) {
  int b = blockIdx.x, i = threadIdx.x;   // blockDim = 256 = D/2
  float t = (float)ts[b];
  float f = __expf(-logf(10000.f) * (float)i / 256.f);
  float a = t * f;
  temb[(long)b * D_ + i]       = cosf(a);
  temb[(long)b * D_ + 256 + i] = sinf(a);
}

__global__ void k_silu(const float* x, float* y, long n) {
  long i = (long)blockIdx.x * blockDim.x + threadIdx.x;
  if (i < n) y[i] = silu_f(x[i]);
}

// h[b*T+t, :] += seq_emb[t, :]
__global__ void k_add_seq(float* h, const float* seq, long n) {
  long i = (long)blockIdx.x * blockDim.x + threadIdx.x;
  if (i < n) h[i] += seq[i % ((long)T_ * D_)];
}

// v[row, :] *= mask[row]
__global__ void k_mask_v(float* v, const float* mask, long n, int dim) {
  long i = (long)blockIdx.x * blockDim.x + threadIdx.x;
  if (i < n) v[i] *= mask[i / dim];
}

// LayerNorm: one wave per row
__global__ void k_ln(const float* x, const float* g, const float* b, float* y,
                     long rows, int dim) {
  long wid = ((long)blockIdx.x * blockDim.x + threadIdx.x) >> 5;
  int lane = threadIdx.x & 31;
  if (wid >= rows) return;
  const float* xr = x + wid * dim;
  float s = 0.f, s2 = 0.f;
  for (int i = lane; i < dim; i += 32) { float v = xr[i]; s += v; s2 += v * v; }
  for (int o = 16; o > 0; o >>= 1) { s += __shfl_xor(s, o, 32); s2 += __shfl_xor(s2, o, 32); }
  float mu = s / dim;
  float inv = rsqrtf(s2 / dim - mu * mu + 1e-5f);
  float* yr = y + wid * dim;
  for (int i = lane; i < dim; i += 32) yr[i] = (xr[i] - mu) * inv * g[i] + b[i];
}

// Softmax over trailing contiguous 64 (per-head dim); one wave per row.
__global__ void k_softmax64(float* x, long rows) {
  long wid = ((long)blockIdx.x * blockDim.x + threadIdx.x) >> 5;
  int lane = threadIdx.x & 31;
  if (wid >= rows) return;
  float* xr = x + wid * 64;
  float v0 = xr[lane], v1 = xr[lane + 32];
  float m = fmaxf(v0, v1);
  for (int o = 16; o > 0; o >>= 1) m = fmaxf(m, __shfl_xor(m, o, 32));
  float e0 = __expf(v0 - m), e1 = __expf(v1 - m);
  float s = e0 + e1;
  for (int o = 16; o > 0; o >>= 1) s += __shfl_xor(s, o, 32);
  float inv = 1.f / s;
  xr[lane] = e0 * inv; xr[lane + 32] = e1 * inv;
}

// Softmax over sequence axis: x[bn, S, dim], softmax over S per (bn, d).
// Optional mask [bn, S]: adds (1-mask)*-1e6 before softmax.
__global__ void k_softmax_seq(float* x, const float* mask, int bn, int S, int dim) {
  long idx = (long)blockIdx.x * blockDim.x + threadIdx.x;
  if (idx >= (long)bn * dim) return;
  int b = (int)(idx / dim), d = (int)(idx % dim);
  float* base = x + (long)b * S * dim + d;
  float m = -1e30f;
  for (int n = 0; n < S; ++n) {
    float v = base[(long)n * dim];
    if (mask) v += (1.f - mask[(long)b * S + n]) * -1e6f;
    m = fmaxf(m, v);
  }
  float s = 0.f;
  for (int n = 0; n < S; ++n) {
    float v = base[(long)n * dim];
    if (mask) v += (1.f - mask[(long)b * S + n]) * -1e6f;
    v = __expf(v - m); s += v; base[(long)n * dim] = v;
  }
  float inv = 1.f / s;
  for (int n = 0; n < S; ++n) base[(long)n * dim] *= inv;
}

// FiLM: out = silu( LN(y)*(1+eo[b,0:D]) + eo[b,D:2D] ); one wave per row.
__global__ void k_film(const float* y, const float* g, const float* b,
                       const float* eo, float* out, long rows, int tdim, int dim) {
  long wid = ((long)blockIdx.x * blockDim.x + threadIdx.x) >> 5;
  int lane = threadIdx.x & 31;
  if (wid >= rows) return;
  int bi = (int)(wid / tdim);
  const float* yr = y + wid * dim;
  float s = 0.f, s2 = 0.f;
  for (int i = lane; i < dim; i += 32) { float v = yr[i]; s += v; s2 += v * v; }
  for (int o = 16; o > 0; o >>= 1) { s += __shfl_xor(s, o, 32); s2 += __shfl_xor(s2, o, 32); }
  float mu = s / dim;
  float inv = rsqrtf(s2 / dim - mu * mu + 1e-5f);
  const float* eb = eo + (long)bi * 2 * dim;
  float* orow = out + wid * dim;
  for (int i = lane; i < dim; i += 32) {
    float nv = (yr[i] - mu) * inv * g[i] + b[i];
    float hv = nv * (1.f + eb[i]) + eb[dim + i];
    orow[i] = silu_f(hv);
  }
}

// ---------------------------------------------------------------------------
// Host-side launch helpers
// ---------------------------------------------------------------------------
static inline void gemm_launch(hipStream_t s,
    const float* A, const float* W, const float* bias, const float* add, float* C,
    int M, int N, int K,
    long a_rs, long a_cs, long w_rs, long w_cs, long c_rs,
    int batch, int zdiv,
    long a_bs0, long a_bs1, long w_bs0, long w_bs1, long c_bs0, long c_bs1,
    int act) {
  GemmP p;
  p.A = A; p.W = W; p.bias = bias; p.add = add; p.C = C;
  p.M = M; p.N = N; p.K = K;
  p.a_rs = a_rs; p.a_cs = a_cs; p.w_rs = w_rs; p.w_cs = w_cs; p.c_rs = c_rs;
  p.a_bs0 = a_bs0; p.a_bs1 = a_bs1; p.w_bs0 = w_bs0; p.w_bs1 = w_bs1;
  p.c_bs0 = c_bs0; p.c_bs1 = c_bs1;
  p.zdiv = zdiv; p.act = act;
  dim3 grid((N + BNt - 1) / BNt, (M + BMt - 1) / BMt, batch);
  gemm_wmma<<<grid, dim3(256), 0, s>>>(p);
}

static inline void gemm_rm(hipStream_t s, const float* A, const float* W,
    const float* bias, const float* add, float* C, int M, int N, int K, int act) {
  gemm_launch(s, A, W, bias, add, C, M, N, K, K, 1, N, 1, N,
              1, 1, 0, 0, 0, 0, 0, 0, act);
}

static inline long cdiv(long a, long b) { return (a + b - 1) / b; }

extern "C" void kernel_launch(void* const* d_in, const int* in_sizes, int n_in,
                              void* d_out, int out_size, void* d_ws, size_t ws_size,
                              hipStream_t stream) {
  (void)in_sizes; (void)n_in; (void)out_size; (void)ws_size;

  const float* x       = (const float*)d_in[0];   // [B,T,FIN]
  const int*   ts      = (const int*)d_in[1];     // [B]
  const float* xf      = (const float*)d_in[2];   // [B,NTXT,DT]
  const float* mask    = (const float*)d_in[3];   // [B,T,1]
  const float* joint_w = (const float*)d_in[4];
  const float* joint_b = (const float*)d_in[5];
  const float* seq_emb = (const float*)d_in[6];
  const float* t_w1    = (const float*)d_in[7];
  const float* t_b1    = (const float*)d_in[8];
  const float* t_w2    = (const float*)d_in[9];
  const float* t_b2    = (const float*)d_in[10];
  const float* out_w   = (const float*)d_in[11];
  const float* out_b   = (const float*)d_in[12];
  auto LP = [&](int i) { return (const float*)d_in[13 + i]; };

  // ---- workspace layout (floats) ----
  float* ws = (float*)d_ws;
  size_t o = 0;
  float* w_temb = ws + o; o += (size_t)B_ * D_;
  float* w_u    = ws + o; o += (size_t)B_ * E_;
  float* w_emb  = ws + o; o += (size_t)B_ * E_;
  float* w_semb = ws + o; o += (size_t)B_ * E_;
  float* w_eo   = ws + o; o += (size_t)B_ * 2 * D_;
  float* w_h    = ws + o; o += (size_t)BT_ * D_;
  float* w_t0   = ws + o; o += (size_t)BT_ * D_;   // xn
  float* w_t1   = ws + o; o += (size_t)BT_ * D_;   // q / film
  float* w_t2   = ws + o; o += (size_t)BT_ * D_;   // k / kx
  float* w_t3   = ws + o; o += (size_t)BT_ * D_;   // v / vx
  float* w_t4   = ws + o; o += (size_t)BT_ * D_;   // y
  float* w_att  = ws + o; o += (size_t)B_ * H_ * DH_ * DH_;
  float* w_tn   = ws + o; o += (size_t)B_ * NTXT_ * DT_;
  float* w_ffc  = ws + o; o += (size_t)CH_ * FF_;

  const int THR = 256;

  // ======== time embedding + MLP ========
  k_temb<<<B_, 256, 0, stream>>>(ts, w_temb);
  gemm_rm(stream, w_temb, t_w1, t_b1, nullptr, w_u, B_, E_, D_, ACT_SILU);
  gemm_rm(stream, w_u, t_w2, t_b2, nullptr, w_emb, B_, E_, E_, ACT_NONE);
  k_silu<<<(int)cdiv((long)B_ * E_, THR), THR, 0, stream>>>(w_emb, w_semb, (long)B_ * E_);

  // ======== input projection + positional embedding ========
  gemm_rm(stream, x, joint_w, joint_b, nullptr, w_h, BT_, D_, FIN_, ACT_NONE);
  k_add_seq<<<(int)cdiv((long)BT_ * D_, THR), THR, 0, stream>>>(w_h, seq_emb, (long)BT_ * D_);

  const int wavesBT = (int)cdiv((long)BT_ * 32, THR);   // wave-per-row grids
  const long bsA = (long)T_ * D_;                       // per-b stride of [B,T,D]
  const long bsX = (long)NTXT_ * D_;                    // per-b stride of [B,77,D]
  const long bsC = (long)H_ * DH_ * DH_;                // per-b stride of att

  for (int l = 0; l < L_; ++l) {
    const float* sa_norm_g = LP(0)  + (size_t)l * D_;
    const float* sa_norm_b = LP(1)  + (size_t)l * D_;
    const float* sa_q_w    = LP(2)  + (size_t)l * D_ * D_;
    const float* sa_q_b    = LP(3)  + (size_t)l * D_;
    const float* sa_k_w    = LP(4)  + (size_t)l * D_ * D_;
    const float* sa_k_b    = LP(5)  + (size_t)l * D_;
    const float* sa_v_w    = LP(6)  + (size_t)l * D_ * D_;
    const float* sa_v_b    = LP(7)  + (size_t)l * D_;
    const float* sa_emb_w  = LP(8)  + (size_t)l * E_ * 2 * D_;
    const float* sa_emb_b  = LP(9)  + (size_t)l * 2 * D_;
    const float* sa_sng    = LP(10) + (size_t)l * D_;
    const float* sa_snb    = LP(11) + (size_t)l * D_;
    const float* sa_out_w  = LP(12) + (size_t)l * D_ * D_;
    const float* sa_out_b  = LP(13) + (size_t)l * D_;
    const float* ca_norm_g = LP(14) + (size_t)l * D_;
    const float* ca_norm_b = LP(15) + (size_t)l * D_;
    const float* ca_tng    = LP(16) + (size_t)l * DT_;
    const float* ca_tnb    = LP(17) + (size_t)l * DT_;
    const float* ca_q_w    = LP(18) + (size_t)l * D_ * D_;
    const float* ca_q_b    = LP(19) + (size_t)l * D_;
    const float* ca_k_w    = LP(20) + (size_t)l * DT_ * D_;
    const float* ca_k_b    = LP(21) + (size_t)l * D_;
    const float* ca_v_w    = LP(22) + (size_t)l * DT_ * D_;
    const float* ca_v_b    = LP(23) + (size_t)l * D_;
    const float* ca_emb_w  = LP(24) + (size_t)l * E_ * 2 * D_;
    const float* ca_emb_b  = LP(25) + (size_t)l * 2 * D_;
    const float* ca_sng    = LP(26) + (size_t)l * D_;
    const float* ca_snb    = LP(27) + (size_t)l * D_;
    const float* ca_out_w  = LP(28) + (size_t)l * D_ * D_;
    const float* ca_out_b  = LP(29) + (size_t)l * D_;
    const float* f1_w      = LP(30) + (size_t)l * D_ * FF_;
    const float* f1_b      = LP(31) + (size_t)l * FF_;
    const float* f2_w      = LP(32) + (size_t)l * FF_ * D_;
    const float* f2_b      = LP(33) + (size_t)l * D_;
    const float* f_emb_w   = LP(34) + (size_t)l * E_ * 2 * D_;
    const float* f_emb_b   = LP(35) + (size_t)l * 2 * D_;
    const float* f_sng     = LP(36) + (size_t)l * D_;
    const float* f_snb     = LP(37) + (size_t)l * D_;
    const float* f_out_w   = LP(38) + (size_t)l * D_ * D_;
    const float* f_out_b   = LP(39) + (size_t)l * D_;

    // ---------------- self attention ----------------
    k_ln<<<wavesBT, THR, 0, stream>>>(w_h, sa_norm_g, sa_norm_b, w_t0, BT_, D_);
    gemm_rm(stream, w_t0, sa_q_w, sa_q_b, nullptr, w_t1, BT_, D_, D_, ACT_NONE);
    gemm_rm(stream, w_t0, sa_k_w, sa_k_b, nullptr, w_t2, BT_, D_, D_, ACT_NONE);
    gemm_rm(stream, w_t0, sa_v_w, sa_v_b, nullptr, w_t3, BT_, D_, D_, ACT_NONE);
    k_mask_v<<<(int)cdiv((long)BT_ * D_, THR), THR, 0, stream>>>(w_t3, mask, (long)BT_ * D_, D_);
    k_softmax64<<<(int)cdiv((long)BT_ * H_ * 32, THR), THR, 0, stream>>>(w_t1, (long)BT_ * H_);
    k_softmax_seq<<<(int)cdiv((long)B_ * D_, THR), THR, 0, stream>>>(w_t2, mask, B_, T_, D_);
    // att[b,h] = k^T v : [64 x 196] @ [196 x 64]  (A transposed via strides)
    gemm_launch(stream, w_t2, w_t3, nullptr, nullptr, w_att, DH_, DH_, T_,
                1, D_, D_, 1, DH_, B_ * H_, H_,
                bsA, DH_, bsA, DH_, bsC, (long)DH_ * DH_, ACT_NONE);
    // y[b,h] = q @ att : [196 x 64] @ [64 x 64]
    gemm_launch(stream, w_t1, w_att, nullptr, nullptr, w_t4, T_, DH_, DH_,
                D_, 1, DH_, 1, D_, B_ * H_, H_,
                bsA, DH_, bsC, (long)DH_ * DH_, bsA, DH_, ACT_NONE);
    // stylization + residual
    gemm_rm(stream, w_semb, sa_emb_w, sa_emb_b, nullptr, w_eo, B_, 2 * D_, E_, ACT_NONE);
    k_film<<<wavesBT, THR, 0, stream>>>(w_t4, sa_sng, sa_snb, w_eo, w_t1, BT_, T_, D_);
    gemm_rm(stream, w_t1, sa_out_w, sa_out_b, w_h, w_h, BT_, D_, D_, ACT_NONE);

    // ---------------- cross attention ----------------
    k_ln<<<wavesBT, THR, 0, stream>>>(w_h, ca_norm_g, ca_norm_b, w_t0, BT_, D_);
    k_ln<<<(int)cdiv((long)B_ * NTXT_ * 32, THR), THR, 0, stream>>>(xf, ca_tng, ca_tnb, w_tn,
                                                                    (long)B_ * NTXT_, DT_);
    gemm_rm(stream, w_t0, ca_q_w, ca_q_b, nullptr, w_t1, BT_, D_, D_, ACT_NONE);
    gemm_rm(stream, w_tn, ca_k_w, ca_k_b, nullptr, w_t2, B_ * NTXT_, D_, DT_, ACT_NONE);
    gemm_rm(stream, w_tn, ca_v_w, ca_v_b, nullptr, w_t3, B_ * NTXT_, D_, DT_, ACT_NONE);
    k_softmax64<<<(int)cdiv((long)BT_ * H_ * 32, THR), THR, 0, stream>>>(w_t1, (long)BT_ * H_);
    k_softmax_seq<<<(int)cdiv((long)B_ * D_, THR), THR, 0, stream>>>(w_t2, nullptr, B_, NTXT_, D_);
    gemm_launch(stream, w_t2, w_t3, nullptr, nullptr, w_att, DH_, DH_, NTXT_,
                1, D_, D_, 1, DH_, B_ * H_, H_,
                bsX, DH_, bsX, DH_, bsC, (long)DH_ * DH_, ACT_NONE);
    gemm_launch(stream, w_t1, w_att, nullptr, nullptr, w_t4, T_, DH_, DH_,
                D_, 1, DH_, 1, D_, B_ * H_, H_,
                bsA, DH_, bsC, (long)DH_ * DH_, bsA, DH_, ACT_NONE);
    gemm_rm(stream, w_semb, ca_emb_w, ca_emb_b, nullptr, w_eo, B_, 2 * D_, E_, ACT_NONE);
    k_film<<<wavesBT, THR, 0, stream>>>(w_t4, ca_sng, ca_snb, w_eo, w_t1, BT_, T_, D_);
    gemm_rm(stream, w_t1, ca_out_w, ca_out_b, w_h, w_h, BT_, D_, D_, ACT_NONE);

    // ---------------- FFN (row-chunked to bound workspace) ----------------
    for (int c = 0; c < BT_ / CH_; ++c) {
      const float* hrow = w_h + (size_t)c * CH_ * D_;
      float* yrow = w_t4 + (size_t)c * CH_ * D_;
      gemm_rm(stream, hrow, f1_w, f1_b, nullptr, w_ffc, CH_, FF_, D_, ACT_GELU);
      gemm_rm(stream, w_ffc, f2_w, f2_b, nullptr, yrow, CH_, D_, FF_, ACT_NONE);
    }
    gemm_rm(stream, w_semb, f_emb_w, f_emb_b, nullptr, w_eo, B_, 2 * D_, E_, ACT_NONE);
    k_film<<<wavesBT, THR, 0, stream>>>(w_t4, f_sng, f_snb, w_eo, w_t1, BT_, T_, D_);
    gemm_rm(stream, w_t1, f_out_w, f_out_b, w_h, w_h, BT_, D_, D_, ACT_NONE);
  }

  // ======== output projection ========
  gemm_rm(stream, w_h, out_w, out_b, nullptr, (float*)d_out, BT_, FIN_, D_, ACT_NONE);
}
